// RecurrentEncoder_32598801776798
// MI455X (gfx1250) — matrix operand order repaired and checked by reading, not compiled, not run on hardware
//
#include <hip/hip_runtime.h>

// ---------------------------------------------------------------------------
// CDNA5 (gfx1250) WMMA recurrent encoder, round 4.
// Key insight: GRU recurrence is independent per batch row -> split B=128
// into 8 groups of 16 rows (one WMMA M-tile each) with NO cross-WG sync.
// Branch scans: 24 WGs (3 branches x 8 batch groups), state 16xH double-
// buffered in static LDS.  Block scans: 8 WGs.  All matmuls via
// v_wmma_f32_16x16x32_f16.
// ---------------------------------------------------------------------------

typedef __attribute__((ext_vector_type(16))) _Float16 v16h;
typedef __attribute__((ext_vector_type(8)))  _Float16 v8h;
typedef __attribute__((ext_vector_type(8)))  float    v8f;

union V16U { v16h v; v8h h2[2]; _Float16 e[16]; };

#define DEVINL __device__ __forceinline__

DEVINL v8f wmma16(v16h a, v16h b, v8f c) {
  return __builtin_amdgcn_wmma_f32_16x16x32_f16(false, a, false, b, (short)0, c,
                                                false, false);
}

// --- A fragment (16x32 f16, row-major, ld elements) ------------------------
// lanes 0-15:  row m0+l, halves [k0..k0+7] and [k0+16..k0+23]
// lanes 16-31: row m0+l, halves [k0+8..k0+15] and [k0+24..k0+31]
DEVINL v16h ldA16(const _Float16* base, int ld, int m0, int k0, int lane) {
  int l = lane & 15, hi = (lane >> 4) & 1;
  const _Float16* p = base + (size_t)(m0 + l) * ld + k0 + hi * 8;
  V16U u;
  u.h2[0] = *(const v8h*)(p);
  u.h2[1] = *(const v8h*)(p + 16);
  return u.v;
}

// --- B fragment (32x16): B[k][n] = W[n][k], W row-major (N,K) f16 ----------
DEVINL v16h ldB16(const _Float16* W, int K, int n0, int k0, int lane) {
  int n = lane & 15, hi = (lane >> 4) & 1;
  const _Float16* p = W + (size_t)(n0 + n) * K + k0 + hi * 8;
  V16U u;
  u.h2[0] = *(const v8h*)(p);
  u.h2[1] = *(const v8h*)(p + 16);
  return u.v;
}

DEVINL float sigm(float x) { return 1.0f / (1.0f + __expf(-x)); }

#define ZERO8 {0.f, 0.f, 0.f, 0.f, 0.f, 0.f, 0.f, 0.f}

// --- packed nearest-upsample store into h_cat ------------------------------
template <int REP>
DEVINL void store_rep(_Float16* dst, _Float16 hf) {
  if constexpr (REP == 1) {
    dst[0] = hf;
  } else if constexpr (REP == 2) {
    union { _Float16 h[2]; unsigned u; } pk;
    pk.h[0] = hf; pk.h[1] = hf;
    *reinterpret_cast<unsigned*>(dst) = pk.u;
  } else {
    union { _Float16 h[4]; unsigned long long u; } pk;
    pk.h[0] = hf; pk.h[1] = hf; pk.h[2] = hf; pk.h[3] = hf;
    *reinterpret_cast<unsigned long long*>(dst) = pk.u;
  }
}

// ---------------------------------------------------------------------------
// Branch GRU scan over one 16-row batch group.  512 threads (16 waves);
// each wave owns TW column tiles; state 16xH double-buffered in LDS.
// ---------------------------------------------------------------------------
template <int IN, int H, int REP, int COFF>
__device__ void branch_scan_body(const _Float16* __restrict__ x16,  // SB x IN
                                 const _Float16* __restrict__ wih,
                                 const _Float16* __restrict__ whh,
                                 const float* __restrict__ bih,
                                 const float* __restrict__ bhh,
                                 _Float16* __restrict__ hcat,
                                 _Float16* smem,  // LDS, 2*16*H f16
                                 int m0 /* batch row base */) {
  constexpr int NCOL = H / 16;
  constexpr int TW = (H >= 512) ? 2 : 1;  // column tiles per wave
  const int tid = threadIdx.x;
  const int lane = tid & 31, wave = tid >> 5;
  const int ln15 = lane & 15, mh = (lane >> 4) * 8;
  const int c0 = wave * TW;
  const bool active = (c0 < NCOL);

  for (int i = tid; i < 16 * H; i += 512) smem[i] = (_Float16)0.0f;
  __syncthreads();

  // time-invariant gate biases per owned column tile
  float br_[TW], bz_[TW], bxn_[TW], bhn_[TW];
#pragma unroll
  for (int t = 0; t < TW; ++t) {
    br_[t] = bz_[t] = bxn_[t] = bhn_[t] = 0.0f;
    if (active) {
      const int j = (c0 + t) * 16 + ln15;
      br_[t]  = bih[j] + bhh[j];
      bz_[t]  = bih[H + j] + bhh[H + j];
      bxn_[t] = bih[2 * H + j];
      bhn_[t] = bhh[2 * H + j];
    }
  }

  for (int s = 0; s < 1024; ++s) {
    _Float16* cur = smem + ((s & 1) ? 16 * H : 0);
    _Float16* nxt = smem + ((s & 1) ? 0 : 16 * H);

    if (active) {
      const _Float16* xs = x16 + (size_t)s * 128 * IN;
      v8f ar[TW], az[TW], axn[TW], ahn[TW];
#pragma unroll
      for (int t = 0; t < TW; ++t) {
        v8f z = ZERO8;
        ar[t] = z; az[t] = z; axn[t] = z; ahn[t] = z;
      }

      // input chains: all tiles share the A fragment (same 16 batch rows)
#pragma unroll
      for (int k = 0; k < IN; k += 32) {
        const v16h a = ldA16(xs, IN, m0, k, lane);
#pragma unroll
        for (int t = 0; t < TW; ++t) {
          const int n0 = (c0 + t) * 16;
          ar[t]  = wmma16(a, ldB16(wih, IN, n0,         k, lane), ar[t]);
          az[t]  = wmma16(a, ldB16(wih, IN, n0 + H,     k, lane), az[t]);
          axn[t] = wmma16(a, ldB16(wih, IN, n0 + 2 * H, k, lane), axn[t]);
        }
      }
      // recurrent chains: state in LDS (local rows 0..15)
#pragma unroll
      for (int k = 0; k < H; k += 32) {
        const v16h a = ldA16(cur, H, 0, k, lane);
#pragma unroll
        for (int t = 0; t < TW; ++t) {
          const int n0 = (c0 + t) * 16;
          ar[t]  = wmma16(a, ldB16(whh, H, n0,         k, lane), ar[t]);
          az[t]  = wmma16(a, ldB16(whh, H, n0 + H,     k, lane), az[t]);
          ahn[t] = wmma16(a, ldB16(whh, H, n0 + 2 * H, k, lane), ahn[t]);
        }
      }

#pragma unroll
      for (int t = 0; t < TW; ++t) {
        const int j = (c0 + t) * 16 + ln15;
        const float br = br_[t], bz = bz_[t], bxn = bxn_[t], bhn = bhn_[t];
#pragma unroll
        for (int i = 0; i < 8; ++i) {
          const int lr = mh + i;  // local row 0..15
          const float hold = (float)cur[lr * H + j];
          const float r  = sigm(ar[t][i] + br);
          const float z  = sigm(az[t][i] + bz);
          const float nv = tanhf(axn[t][i] + bxn + r * (ahn[t][i] + bhn));
          const float hn = (1.0f - z) * nv + z * hold;
          nxt[lr * H + j] = (_Float16)hn;
          store_rep<REP>(hcat + ((size_t)s * 128 + m0 + lr) * 1536 + COFF +
                             (size_t)REP * j,
                         (_Float16)hn);
        }
      }
    }
    __syncthreads();
  }
}

__global__ __launch_bounds__(512) void branch_scan_kernel(
    const _Float16* x0, const _Float16* x1, const _Float16* x2,
    const _Float16* w0i, const _Float16* w0h, const float* b0i, const float* b0h,
    const _Float16* w1i, const _Float16* w1h, const float* b1i, const float* b1h,
    const _Float16* w2i, const _Float16* w2h, const float* b2i, const float* b2h,
    _Float16* hcat) {
  __shared__ _Float16 smem[2 * 16 * 512];  // 32 KB, sized for H=512
  const int branch = blockIdx.x >> 3;
  const int m0 = (blockIdx.x & 7) * 16;
  if (branch == 0)
    branch_scan_body<256, 512, 1, 0>(x0, w0i, w0h, b0i, b0h, hcat, smem, m0);
  else if (branch == 1)
    branch_scan_body<128, 256, 2, 512>(x1, w1i, w1h, b1i, b1h, hcat, smem, m0);
  else
    branch_scan_body<64, 128, 4, 1024>(x2, w2i, w2h, b2i, b2h, hcat, smem, m0);
}

// ---------------------------------------------------------------------------
// Residual block GRU scan over one 16-row batch group.  256 threads
// (8 waves = 8 column tiles); g-state 16x128 double-buffered in LDS;
// h[s] += gru(ln[s]) in place.
// ---------------------------------------------------------------------------
__global__ __launch_bounds__(256) void block_scan_kernel(
    const _Float16* __restrict__ ln, const _Float16* __restrict__ wih,
    const _Float16* __restrict__ whh, const float* __restrict__ bih,
    const float* __restrict__ bhh, float* __restrict__ h) {
  constexpr int H = 128, IN = 128;
  __shared__ _Float16 gbuf[2 * 16 * H];  // 8 KB
  const int tid = threadIdx.x, lane = tid & 31, wave = tid >> 5;
  const int ln15 = lane & 15, mh = (lane >> 4) * 8;
  const int m0 = blockIdx.x * 16;
  const int n0 = wave * 16;
  const int j = n0 + ln15;

  for (int i = tid; i < 16 * H; i += 256) gbuf[i] = (_Float16)0.0f;
  __syncthreads();

  const float br  = bih[j] + bhh[j];
  const float bz  = bih[H + j] + bhh[H + j];
  const float bxn = bih[2 * H + j];
  const float bhn = bhh[2 * H + j];

  for (int s = 0; s < 1024; ++s) {
    _Float16* cur = gbuf + ((s & 1) ? 16 * H : 0);
    _Float16* nxt = gbuf + ((s & 1) ? 0 : 16 * H);
    const _Float16* lns = ln + (size_t)s * 128 * IN;
    float* hs = h + (size_t)s * 128 * H;

    v8f ar = ZERO8, az = ZERO8, axn = ZERO8, ahn = ZERO8;

#pragma unroll
    for (int k = 0; k < IN; k += 32) {
      const v16h a = ldA16(lns, IN, m0, k, lane);
      ar  = wmma16(a, ldB16(wih, IN, n0,         k, lane), ar);
      az  = wmma16(a, ldB16(wih, IN, n0 + H,     k, lane), az);
      axn = wmma16(a, ldB16(wih, IN, n0 + 2 * H, k, lane), axn);
    }
#pragma unroll
    for (int k = 0; k < H; k += 32) {
      const v16h a = ldA16(cur, H, 0, k, lane);
      ar  = wmma16(a, ldB16(whh, H, n0,         k, lane), ar);
      az  = wmma16(a, ldB16(whh, H, n0 + H,     k, lane), az);
      ahn = wmma16(a, ldB16(whh, H, n0 + 2 * H, k, lane), ahn);
    }

#pragma unroll
    for (int i = 0; i < 8; ++i) {
      const int lr = mh + i;
      const float gold = (float)cur[lr * H + j];
      const float r  = sigm(ar[i] + br);
      const float z  = sigm(az[i] + bz);
      const float nv = tanhf(axn[i] + bxn + r * (ahn[i] + bhn));
      const float gn = (1.0f - z) * nv + z * gold;
      nxt[lr * H + j] = (_Float16)gn;
      hs[(size_t)(m0 + lr) * H + j] += gn;
    }
    __syncthreads();
  }
}

// ---------------------------------------------------------------------------
// d0 dense: out = relu(hcat(131072x1536,f16) @ W^T(128x1536,f16) + b), f32.
// WG = 128 rows; 8 waves, each a 16-row strip x 8 col tiles.
// ---------------------------------------------------------------------------
__global__ __launch_bounds__(256) void d0_gemm_kernel(
    const _Float16* __restrict__ A, const _Float16* __restrict__ W,
    const float* __restrict__ bias, float* __restrict__ out) {
  const int lane = threadIdx.x & 31, wave = threadIdx.x >> 5;
  const size_t m0 = (size_t)blockIdx.x * 128 + (size_t)wave * 16;
  const _Float16* Arow = A + m0 * 1536;

  v8f acc[8];
#pragma unroll
  for (int c = 0; c < 8; ++c) { v8f z = ZERO8; acc[c] = z; }

  for (int k = 0; k < 1536; k += 32) {
    const v16h a = ldA16(Arow, 1536, 0, k, lane);
#pragma unroll
    for (int c = 0; c < 8; ++c)
      acc[c] = wmma16(a, ldB16(W, 1536, c * 16, k, lane), acc[c]);
  }

  const int n = lane & 15, mh = (lane >> 4) * 8;
#pragma unroll
  for (int c = 0; c < 8; ++c) {
    const int j = c * 16 + n;
    const float bj = bias[j];
#pragma unroll
    for (int i = 0; i < 8; ++i) {
      float v = acc[c][i] + bj;
      out[(m0 + mh + i) * 128 + j] = v > 0.0f ? v : 0.0f;
    }
  }
}

// --- LayerNorm over last dim (128) -> f16 ----------------------------------
__global__ __launch_bounds__(128) void ln_kernel(const float* __restrict__ h,
                                                 const float* __restrict__ g,
                                                 const float* __restrict__ b,
                                                 _Float16* __restrict__ out) {
  __shared__ float red[128];
  const size_t row = blockIdx.x;
  const int j = threadIdx.x;
  const float v = h[row * 128 + j];
  red[j] = v;
  __syncthreads();
  for (int o = 64; o > 0; o >>= 1) {
    if (j < o) red[j] += red[j + o];
    __syncthreads();
  }
  const float m = red[0] * (1.0f / 128.0f);
  __syncthreads();
  const float d = v - m;
  red[j] = d * d;
  __syncthreads();
  for (int o = 64; o > 0; o >>= 1) {
    if (j < o) red[j] += red[j + o];
    __syncthreads();
  }
  const float var = red[0] * (1.0f / 128.0f);
  out[row * 128 + j] = (_Float16)(d * rsqrtf(var + 1e-5f) * g[j] + b[j]);
}

// --- head: 128 -> 4 --------------------------------------------------------
__global__ __launch_bounds__(256) void dense_out_kernel(
    const float* __restrict__ h, const float* __restrict__ w,
    const float* __restrict__ b, float* __restrict__ out) {
  const size_t idx = (size_t)blockIdx.x * 256 + threadIdx.x;
  const size_t row = idx >> 2;
  const int o = (int)(idx & 3);
  const float* hr = h + row * 128;
  const float* wr = w + o * 128;
  float acc = b[o];
#pragma unroll 8
  for (int k = 0; k < 128; ++k) acc += hr[k] * wr[k];
  out[row * 4 + o] = acc;
}

// --- f32 -> f16 conversions ------------------------------------------------
__global__ void cvt_kernel(const float* __restrict__ src,
                           _Float16* __restrict__ dst, int n) {
  int i = blockIdx.x * 256 + threadIdx.x;
  if (i < n) dst[i] = (_Float16)src[i];
}

// strided feature downsample + convert: dst[r][c] = src[r*256 + c*st]
__global__ void cvt_ds_kernel(const float* __restrict__ src,
                              _Float16* __restrict__ dst, int shift, int st,
                              long long total) {
  long long idx = (long long)blockIdx.x * 256 + threadIdx.x;
  if (idx >= total) return;
  long long r = idx >> shift;
  int c = (int)(idx & ((1 << shift) - 1));
  dst[idx] = (_Float16)src[r * 256 + (long long)c * st];
}

// ---------------------------------------------------------------------------
extern "C" void kernel_launch(void* const* d_in, const int* in_sizes, int n_in,
                              void* d_out, int out_size, void* d_ws,
                              size_t ws_size, hipStream_t stream) {
  const float* x        = (const float*)d_in[0];
  const float* g0_wih   = (const float*)d_in[1];
  const float* g0_whh   = (const float*)d_in[2];
  const float* g0_bih   = (const float*)d_in[3];
  const float* g0_bhh   = (const float*)d_in[4];
  const float* g1_wih   = (const float*)d_in[5];
  const float* g1_whh   = (const float*)d_in[6];
  const float* g1_bih   = (const float*)d_in[7];
  const float* g1_bhh   = (const float*)d_in[8];
  const float* g2_wih   = (const float*)d_in[9];
  const float* g2_whh   = (const float*)d_in[10];
  const float* g2_bih   = (const float*)d_in[11];
  const float* g2_bhh   = (const float*)d_in[12];
  const float* d0_w     = (const float*)d_in[13];
  const float* d0_b     = (const float*)d_in[14];
  const float* b0_ln_g  = (const float*)d_in[15];
  const float* b0_ln_b  = (const float*)d_in[16];
  const float* b0_wih   = (const float*)d_in[17];
  const float* b0_whh   = (const float*)d_in[18];
  const float* b0_bih   = (const float*)d_in[19];
  const float* b0_bhh   = (const float*)d_in[20];
  const float* b1_ln_g  = (const float*)d_in[21];
  const float* b1_ln_b  = (const float*)d_in[22];
  const float* b1_wih   = (const float*)d_in[23];
  const float* b1_whh   = (const float*)d_in[24];
  const float* b1_bih   = (const float*)d_in[25];
  const float* b1_bhh   = (const float*)d_in[26];
  const float* df_w     = (const float*)d_in[27];
  const float* df_b     = (const float*)d_in[28];

  const size_t SB = 1024 * 128;  // 131072 rows

  size_t off = 0;
  auto take = [&](size_t bytes) -> void* {
    off = (off + 255) & ~(size_t)255;
    void* p = (char*)d_ws + off;
    off += bytes;
    return p;
  };
  _Float16* hcat  = (_Float16*)take(SB * 1536 * sizeof(_Float16));  // 402 MB
  float*    hbuf  = (float*)take(SB * 128 * sizeof(float));         //  67 MB
  _Float16* lnbuf = (_Float16*)take(SB * 128 * sizeof(_Float16));   //  34 MB
  _Float16* x16_0 = (_Float16*)take(SB * 256 * sizeof(_Float16));   //  67 MB
  _Float16* x16_1 = (_Float16*)take(SB * 128 * sizeof(_Float16));   //  34 MB
  _Float16* x16_2 = (_Float16*)take(SB * 64 * sizeof(_Float16));    //  17 MB

  auto take16 = [&](size_t n) -> _Float16* {
    return (_Float16*)take(n * sizeof(_Float16));
  };
  _Float16* w0i  = take16(1536 * 256);
  _Float16* w0h  = take16(1536 * 512);
  _Float16* w1i  = take16(768 * 128);
  _Float16* w1h  = take16(768 * 256);
  _Float16* w2i  = take16(384 * 64);
  _Float16* w2h  = take16(384 * 128);
  _Float16* wd0  = take16(128 * 1536);
  _Float16* wb0i = take16(384 * 128);
  _Float16* wb0h = take16(384 * 128);
  _Float16* wb1i = take16(384 * 128);
  _Float16* wb1h = take16(384 * 128);

  auto cvt = [&](const float* s, _Float16* d, int n) {
    cvt_kernel<<<(n + 255) / 256, 256, 0, stream>>>(s, d, n);
  };
  cvt(g0_wih, w0i, 1536 * 256);
  cvt(g0_whh, w0h, 1536 * 512);
  cvt(g1_wih, w1i, 768 * 128);
  cvt(g1_whh, w1h, 768 * 256);
  cvt(g2_wih, w2i, 384 * 64);
  cvt(g2_whh, w2h, 384 * 128);
  cvt(d0_w,   wd0, 128 * 1536);
  cvt(b0_wih, wb0i, 384 * 128);
  cvt(b0_whh, wb0h, 384 * 128);
  cvt(b1_wih, wb1i, 384 * 128);
  cvt(b1_whh, wb1h, 384 * 128);

  // dense f16 copies of x at the three feature resolutions
  {
    long long t0 = (long long)SB * 256;
    cvt_ds_kernel<<<(unsigned)((t0 + 255) / 256), 256, 0, stream>>>(
        x, x16_0, 8, 1, t0);
    long long t1 = (long long)SB * 128;
    cvt_ds_kernel<<<(unsigned)((t1 + 255) / 256), 256, 0, stream>>>(
        x, x16_1, 7, 2, t1);
    long long t2 = (long long)SB * 64;
    cvt_ds_kernel<<<(unsigned)((t2 + 255) / 256), 256, 0, stream>>>(
        x, x16_2, 6, 4, t2);
  }

  // --- branch scans: 3 branches x 8 independent batch groups = 24 WGs ---
  branch_scan_kernel<<<24, 512, 0, stream>>>(
      x16_0, x16_1, x16_2, w0i, w0h, g0_bih, g0_bhh, w1i, w1h, g1_bih, g1_bhh,
      w2i, w2h, g2_bih, g2_bhh, hcat);

  // --- d0 dense + relu ---
  d0_gemm_kernel<<<SB / 128, 256, 0, stream>>>(hcat, wd0, d0_b, hbuf);

  // --- residual blocks (8 independent batch groups each) ---
  ln_kernel<<<(unsigned)SB, 128, 0, stream>>>(hbuf, b0_ln_g, b0_ln_b, lnbuf);
  block_scan_kernel<<<8, 256, 0, stream>>>(lnbuf, wb0i, wb0h, b0_bih, b0_bhh,
                                           hbuf);
  ln_kernel<<<(unsigned)SB, 128, 0, stream>>>(hbuf, b1_ln_g, b1_ln_b, lnbuf);
  block_scan_kernel<<<8, 256, 0, stream>>>(lnbuf, wb1i, wb1h, b1_bih, b1_bhh,
                                           hbuf);

  // --- head: 128 -> 4 ---
  dense_out_kernel<<<(unsigned)(SB * 4 / 256), 256, 0, stream>>>(
      hbuf, df_w, df_b, (float*)d_out);
}
